// CausalMultiheadAttention_82832739270861
// MI455X (gfx1250) — matrix operand order repaired and checked by reading, not compiled, not run on hardware
//
#include <hip/hip_runtime.h>
#include <math.h>

// ---------------------------------------------------------------------------
// CDNA5 / gfx1250 causal multi-head attention forward.
// B=4, S=2048, D_MODEL=1024, H=16, D_K=64.
// All matmuls on v_wmma_f32_16x16x32_bf16 (wave32). Double-buffered LDS,
// ds_swizzle_b32 softmax reductions, softmax scale folded into Q.
// ---------------------------------------------------------------------------

typedef __attribute__((ext_vector_type(16))) __bf16 v16bf;
typedef __attribute__((ext_vector_type(8)))  float  v8f;

#define DEVINL __device__ __forceinline__

// -------------------- WMMA fragment helpers (ISA 7.12.2 layouts) -----------

// A-matrix 16x32 bf16 fragment from a row-major tile (row stride in halfs).
// lane<16 : row=lane,    halves = K[0..7]  ++ K[16..23]
// lane>=16: row=lane-16, halves = K[8..15] ++ K[24..31]
DEVINL v16bf load_frag_a(const __bf16* base, int stride, int lane) {
  const int r  = lane & 15;
  const int h8 = (lane >> 4) << 3;
  const __bf16* row = base + r * stride + h8;
  union { v16bf v; float4 f[2]; } u;
  u.f[0] = *reinterpret_cast<const float4*>(row);
  u.f[1] = *reinterpret_cast<const float4*>(row + 16);
  return u.v;
}

// B-matrix 32x16 bf16 fragment, sourced from Bt[n][k] (= W rows), row-major.
// lane<16 : col n=lane, halves = K[0..7] ++ K[16..23]; lane>=16: K[8..15] ++ K[24..31].
DEVINL v16bf load_frag_b(const __bf16* base, int stride, int lane) {
  const int n   = lane & 15;
  const int h16 = (lane >> 4) << 3;
  const __bf16* row = base + n * stride + h16;
  const float4* p = reinterpret_cast<const float4*>(row);
  union { v16bf v; float4 f[2]; } u;
  u.f[0] = p[0];
  u.f[1] = p[2];
  return u.v;
}

DEVINL v8f wmma_bf16(v16bf a, v16bf b, v8f c) {
  return __builtin_amdgcn_wmma_f32_16x16x32_bf16(false, a, false, b,
                                                 (short)0, c, false, false);
}

// ds_swizzle_b32 lane-xor exchange (group-of-32 mode: and=0x1f, or=0, xor=M).
// Masks 1/2/4/8 stay inside each 16-lane half that owns a C-layout row.
template <int M>
DEVINL float ds_swz_xor(float v) {
  int i = __builtin_amdgcn_ds_swizzle(__float_as_int(v), (M << 10) | 0x1f);
  return __int_as_float(i);
}

DEVINL float half_max16(float v) {
  v = fmaxf(v, ds_swz_xor<1>(v));
  v = fmaxf(v, ds_swz_xor<2>(v));
  v = fmaxf(v, ds_swz_xor<4>(v));
  v = fmaxf(v, ds_swz_xor<8>(v));
  return v;
}

DEVINL float half_sum16(float v) {
  v += ds_swz_xor<1>(v);
  v += ds_swz_xor<2>(v);
  v += ds_swz_xor<4>(v);
  v += ds_swz_xor<8>(v);
  return v;
}

// -------------------- fp32 -> bf16 conversion ------------------------------

__global__ __launch_bounds__(256) void f32_to_bf16(const float* __restrict__ in,
                                                   __bf16* __restrict__ out,
                                                   int n) {
  int i = blockIdx.x * blockDim.x + threadIdx.x;
  if (i < n) out[i] = (__bf16)in[i];
}

// -------------------- GEMM: out(m,n) = sum_k A[m][k] * W[n][k] -------------
// Block tile 256(M) x 64(N), K-step 32, 256 threads = 8 waves. Each wave owns
// two 16-row strips (reusing B-fragments): 8 WMMAs per K-step per wave.
// MODE 0: scatter bf16 into qkv[q][b][h][s][64]; Q plane pre-scaled by 0.125.
// MODE 1: write fp32 row-major [M][1024] to d_out.

template <int MODE>
__global__ __launch_bounds__(256)
void gemm_xwt(const __bf16* __restrict__ A, const __bf16* __restrict__ W,
              void* __restrict__ outp, int K) {
  __shared__ __bf16 lA[2][256 * 32];  // 16 KB x2
  __shared__ __bf16 lB[2][64 * 32];   //  4 KB x2

  const int tid  = threadIdx.x;
  const int lane = tid & 31;
  const int wave = tid >> 5;
  const int m0 = blockIdx.y * 256;
  const int n0 = blockIdx.x * 64;

  v8f acc[2][4];
#pragma unroll
  for (int mh = 0; mh < 2; ++mh)
#pragma unroll
    for (int i = 0; i < 4; ++i)
      acc[mh][i] = v8f{0.f,0.f,0.f,0.f,0.f,0.f,0.f,0.f};

  // A: thread t copies its whole 32-half row segment (64 B).
  const __bf16* aptr = A + (size_t)(m0 + tid) * K;
  // B: 64 rows x 32 halfs = 2048 halfs; 8 halfs (16 B) per thread.
  const int brow = tid >> 2, bcol = (tid & 3) * 8;
  const __bf16* bptr = W + (size_t)(n0 + brow) * K + bcol;

  for (int kk = 0; kk < K; kk += 32) {
    const int p = (kk >> 5) & 1;
    {
      float4 a0 = *reinterpret_cast<const float4*>(aptr + kk);
      float4 a1 = *reinterpret_cast<const float4*>(aptr + kk + 8);
      float4 a2 = *reinterpret_cast<const float4*>(aptr + kk + 16);
      float4 a3 = *reinterpret_cast<const float4*>(aptr + kk + 24);
      float4* dst = reinterpret_cast<float4*>(&lA[p][tid * 32]);
      dst[0] = a0; dst[1] = a1; dst[2] = a2; dst[3] = a3;
      float4 bq = *reinterpret_cast<const float4*>(bptr + kk);
      *reinterpret_cast<float4*>(&lB[p][brow * 32 + bcol]) = bq;
      if (kk + 32 < K) {
        __builtin_prefetch(aptr + kk + 32, 0, 0);
        __builtin_prefetch(bptr + kk + 32, 0, 0);
      }
    }
    __syncthreads();   // single barrier per step; WMMA dscnt waits make the
                       // buffer-(p) reads complete before the next arrival.
    v16bf af0 = load_frag_a(&lA[p][(wave * 16) * 32], 32, lane);
    v16bf af1 = load_frag_a(&lA[p][(128 + wave * 16) * 32], 32, lane);
#pragma unroll
    for (int nt = 0; nt < 4; ++nt) {
      v16bf bf = load_frag_b(&lB[p][nt * 16 * 32], 32, lane);
      acc[0][nt] = wmma_bf16(af0, bf, acc[0][nt]);
      acc[1][nt] = wmma_bf16(af1, bf, acc[1][nt]);
    }
  }

  const int rbase = (lane >> 4) * 8;
  const int nlo   = lane & 15;
#pragma unroll
  for (int mh = 0; mh < 2; ++mh) {
#pragma unroll
    for (int nt = 0; nt < 4; ++nt) {
#pragma unroll
      for (int r = 0; r < 8; ++r) {
        float v = acc[mh][nt][r];
        int gm = m0 + mh * 128 + wave * 16 + rbase + r;
        int gn = n0 + nt * 16 + nlo;
        if constexpr (MODE == 0) {
          __bf16* out = (__bf16*)outp;
          int q = gn >> 10, rem = gn & 1023, h = rem >> 6, k = rem & 63;
          int bb = gm >> 11, s = gm & 2047;
          if (q == 0) v *= 0.125f;   // fold 1/sqrt(d_k) into Q (exact in bf16)
          size_t idx = (size_t)q * 8388608 + (size_t)bb * 2097152 +
                       (size_t)h * 131072 + (size_t)s * 64 + k;
          out[idx] = (__bf16)v;
        } else {
          float* out = (float*)outp;
          out[(size_t)gm * 1024 + gn] = v;
        }
      }
    }
  }
}

// -------------------- Flash attention (causal, online softmax) -------------
// One block per (query-block of 128, head, batch). 8 waves, each wave owns
// 16 query rows. Streams key/value blocks of 32, double-buffered in LDS.

__global__ __launch_bounds__(256)
void flash_attn(const __bf16* __restrict__ qkv, __bf16* __restrict__ attn) {
  __shared__ __bf16 lK[2][32 * 64];     // K block row-major [t][k]   4 KB x2
  __shared__ __bf16 lVt[2][64 * 32];    // V block transposed [k][t]  4 KB x2
  __shared__ __bf16 lP[8 * 16 * 32];    // per-wave P staging         8 KB

  const int tid  = threadIdx.x;
  const int lane = tid & 31;
  const int wave = tid >> 5;
  const int qb = blockIdx.x;   // 0..15
  const int h  = blockIdx.y;   // 0..15
  const int b  = blockIdx.z;   // 0..3

  const size_t headoff = ((size_t)b * 16 + h) * (size_t)(2048 * 64);
  const __bf16* Qh = qkv + headoff;                       // pre-scaled by 1/8
  const __bf16* Kh = qkv + (size_t)8388608 + headoff;
  const __bf16* Vh = qkv + (size_t)2 * 8388608 + headoff;

  const int q0 = qb * 128 + wave * 16;

  // Q fragments for both K-steps (d_k = 64 = 2 x 32), A-fragment order.
  v16bf qf[2];
  {
    const int r  = lane & 15;
    const int h8 = (lane >> 4) << 3;
    const __bf16* qrow = Qh + (size_t)(q0 + r) * 64;
#pragma unroll
    for (int ks = 0; ks < 2; ++ks) {
      union { v16bf v; float4 f[2]; } u;
      u.f[0] = *reinterpret_cast<const float4*>(qrow + ks * 32 + h8);
      u.f[1] = *reinterpret_cast<const float4*>(qrow + ks * 32 + h8 + 16);
      qf[ks] = u.v;
    }
  }

  v8f o[4];
#pragma unroll
  for (int i = 0; i < 4; ++i) o[i] = v8f{0.f,0.f,0.f,0.f,0.f,0.f,0.f,0.f};
  float mrow[8], lrow[8];
#pragma unroll
  for (int r = 0; r < 8; ++r) { mrow[r] = -1e30f; lrow[r] = 0.f; }

  const int rbase = (lane >> 4) * 8;
  const int nlo   = lane & 15;
  const int ldrow = tid >> 3;          // 0..31  (K/V row)
  const int ldseg = (tid & 7) * 8;     // 8-half segment within row

  const int kb_end = 4 * (qb + 1);     // causal: keys up to qb*128+127

  for (int kb = 0; kb < kb_end; ++kb) {
    const int p = kb & 1;
    {
      const size_t gr = (size_t)(kb * 32 + ldrow) * 64 + ldseg;
      float4 kt = *reinterpret_cast<const float4*>(Kh + gr);
      *reinterpret_cast<float4*>(&lK[p][ldrow * 64 + ldseg]) = kt;
      float4 vt = *reinterpret_cast<const float4*>(Vh + gr);
      const __bf16* e = reinterpret_cast<const __bf16*>(&vt);
#pragma unroll
      for (int i = 0; i < 8; ++i) lVt[p][(ldseg + i) * 32 + ldrow] = e[i];
    }
    __syncthreads();

    // S = Q . K^T for this wave's 16 rows x 32 keys (2 N-tiles, 2 K-steps)
    v8f s0 = v8f{0.f,0.f,0.f,0.f,0.f,0.f,0.f,0.f};
    v8f s1 = v8f{0.f,0.f,0.f,0.f,0.f,0.f,0.f,0.f};
#pragma unroll
    for (int ks = 0; ks < 2; ++ks) {
      v16bf b0 = load_frag_b(&lK[p][ks * 32], 64, lane);
      s0 = wmma_bf16(qf[ks], b0, s0);
      v16bf b1 = load_frag_b(&lK[p][16 * 64 + ks * 32], 64, lane);
      s1 = wmma_bf16(qf[ks], b1, s1);
    }

    // Online softmax; scale already folded into Q.
    __bf16* lPw = lP + wave * 512;
    const int colg = kb * 32 + nlo;
#pragma unroll
    for (int r = 0; r < 8; ++r) {
      const int rowg = q0 + rbase + r;
      float v0 = (colg      <= rowg) ? s0[r] : -1e30f;
      float v1 = (colg + 16 <= rowg) ? s1[r] : -1e30f;
      float mx = half_max16(fmaxf(v0, v1));
      mx = fmaxf(mx, mrow[r]);
      float alpha = __expf(mrow[r] - mx);
      float p0 = __expf(v0 - mx);
      float p1 = __expf(v1 - mx);
      float rs = half_sum16(p0 + p1);
      lrow[r] = lrow[r] * alpha + rs;
      mrow[r] = mx;
#pragma unroll
      for (int nt = 0; nt < 4; ++nt) o[nt][r] *= alpha;
      // C-layout -> LDS (row-major 16x32); same-wave DS ops are in-order.
      lPw[(rbase + r) * 32 + nlo]      = (__bf16)p0;
      lPw[(rbase + r) * 32 + 16 + nlo] = (__bf16)p1;
    }

    // O += P . V  (A = P 16x32, Bt = Vt[k][t])
    v16bf pf = load_frag_a(lPw, 32, lane);
#pragma unroll
    for (int nt = 0; nt < 4; ++nt) {
      v16bf vb = load_frag_b(&lVt[p][nt * 16 * 32], 32, lane);
      o[nt] = wmma_bf16(pf, vb, o[nt]);
    }
  }

  // Normalize and write attn_cat[b][s][h*64 + k] as bf16.
#pragma unroll
  for (int nt = 0; nt < 4; ++nt) {
#pragma unroll
    for (int r = 0; r < 8; ++r) {
      int srow = q0 + rbase + r;
      int col  = h * 64 + nt * 16 + nlo;
      attn[((size_t)b * 2048 + srow) * 1024 + col] = (__bf16)(o[nt][r] / lrow[r]);
    }
  }
}

// -------------------- host-side orchestration ------------------------------

extern "C" void kernel_launch(void* const* d_in, const int* in_sizes, int n_in,
                              void* d_out, int out_size, void* d_ws, size_t ws_size,
                              hipStream_t stream) {
  const float* x     = (const float*)d_in[0];   // [4,2048,1024]
  const float* w_qkv = (const float*)d_in[1];   // [3,16,64,1024]
  const float* w_o   = (const float*)d_in[2];   // [1024,1024]
  float* out = (float*)d_out;                   // [4,2048,1024]

  char* ws = (char*)d_ws;
  const size_t N_X   = (size_t)8192 * 1024;     // 8,388,608
  const size_t N_WQ  = (size_t)3072 * 1024;     // 3,145,728
  const size_t N_WO  = (size_t)1024 * 1024;     // 1,048,576
  const size_t N_QKV = (size_t)3 * 8388608;     // 25,165,824

  __bf16* xb    = (__bf16*)ws;                      ws += N_X   * 2;
  __bf16* wqkvb = (__bf16*)ws;                      ws += N_WQ  * 2;
  __bf16* wob   = (__bf16*)ws;                      ws += N_WO  * 2;
  __bf16* qkv   = (__bf16*)ws;                      ws += N_QKV * 2;
  __bf16* attn  = (__bf16*)ws;                      // N_X bf16

  f32_to_bf16<<<(int)((N_X  + 255) / 256), 256, 0, stream>>>(x,     xb,    (int)N_X);
  f32_to_bf16<<<(int)((N_WQ + 255) / 256), 256, 0, stream>>>(w_qkv, wqkvb, (int)N_WQ);
  f32_to_bf16<<<(int)((N_WO + 255) / 256), 256, 0, stream>>>(w_o,   wob,   (int)N_WO);

  // QKV projection: [8192,1024] x [3072,1024]^T -> qkv planes (bf16, Q/8)
  gemm_xwt<0><<<dim3(3072 / 64, 8192 / 256), 256, 0, stream>>>(xb, wqkvb, qkv, 1024);

  // Causal flash attention per (query-block, head, batch)
  flash_attn<<<dim3(16, 16, 4), 256, 0, stream>>>(qkv, attn);

  // Output projection: [8192,1024] x [1024,1024]^T -> fp32 d_out
  gemm_xwt<1><<<dim3(1024 / 64, 8192 / 256), 256, 0, stream>>>(attn, wob, out, 1024);
}
